// JKConv_68590627717671
// MI455X (gfx1250) — matrix operand order repaired and checked by reading, not compiled, not run on hardware
//
#include <hip/hip_runtime.h>
#include <math.h>

typedef __attribute__((ext_vector_type(16))) __bf16 v16bf;
typedef __attribute__((ext_vector_type(8)))  float  v8f;

#define D 128   // feature dim (in == out == 128)

// float -> bf16 bits, round-to-nearest-even
__device__ __forceinline__ unsigned short f2bf(float f) {
    union { float f; unsigned u; } v; v.f = f;
    unsigned r = v.u + 0x7FFFu + ((v.u >> 16) & 1u);
    return (unsigned short)(r >> 16);
}

// ---------------------------------------------------------------- utilities
__global__ void fill_kernel(float* p, float v, int n) {
    int i = blockIdx.x * blockDim.x + threadIdx.x;
    if (i < n) p[i] = v;
}

__global__ void deg_kernel(const int* __restrict__ dst, float* deg, int E) {
    int i = blockIdx.x * blockDim.x + threadIdx.x;
    if (i < E) atomicAdd(&deg[dst[i]], 1.0f);
}

__global__ void dinv_kernel(float* deg, int n) {
    int i = blockIdx.x * blockDim.x + threadIdx.x;
    if (i < n) {
        float d = deg[i];
        deg[i] = (d > 0.0f) ? rsqrtf(d) : 0.0f;
    }
}

// convert f32 x -> bf16 Xbf (row-major), 4 elems / thread
__global__ void x2bf_kernel(const float* __restrict__ x,
                            unsigned short* __restrict__ xbf, int total4) {
    int i = blockIdx.x * blockDim.x + threadIdx.x;
    if (i >= total4) return;
    float4 v = ((const float4*)x)[i];
    uint2 o;
    o.x = (unsigned)f2bf(v.x) | ((unsigned)f2bf(v.y) << 16);
    o.y = (unsigned)f2bf(v.z) | ((unsigned)f2bf(v.w) << 16);
    ((uint2*)xbf)[i] = o;
}

// ---------------------------------------------------------------- W swizzle
// Wz[((ks*8 + nt)*32 + lane)*16 + j] = bf16( W[ks*32 + (lane>>4)*16 + j][nt*16 + (lane&15)] )
// => each B fragment is 32 contiguous bytes per lane; GEMM loads it with b128 loads.
__global__ __launch_bounds__(256) void swizzleW_kernel(const float* __restrict__ W,
                                                       unsigned short* __restrict__ Wz) {
    int t = blockIdx.x * blockDim.x + threadIdx.x;   // 1024 (group,lane) pairs
    if (t >= 1024) return;
    int g = t >> 5, lane = t & 31;
    int ks = g >> 3, nt = g & 7;
    int m = lane & 15, half = lane >> 4;
    int col = nt * 16 + m;
    int kb = ks * 32 + half * 16;
    unsigned short o[16];
    #pragma unroll
    for (int j = 0; j < 16; ++j) o[j] = f2bf(W[(kb + j) * D + col]);
    uint4* dst = (uint4*)(Wz + t * 16);
    dst[0] = *(uint4*)(o);
    dst[1] = *(uint4*)(o + 8);
}

// ---------------------------------------------------------------- GEMM (BF16 WMMA)
// H[n x 128] = Xbf[n x 128](bf16) @ Wz(bf16, pre-swizzled), f32 out.
// Block = 256 threads = 8 waves; each wave: 16 rows x 128 cols strip.
// No LDS: A = 2x16B loads/K-step, B fragments = direct b128 loads in WMMA layout.
__global__ __launch_bounds__(256) void gemm_wmma(const unsigned short* __restrict__ Xbf,
                                                 const unsigned short* __restrict__ Wz,
                                                 float* __restrict__ H, int n) {
    const int tid  = threadIdx.x;
    const int wave = tid >> 5;
    const int lane = tid & 31;
    const int m    = lane & 15;
    const int half = lane >> 4;

    const int rowBase = blockIdx.x * 128 + wave * 16;
    // Clamp (not mask) so EXEC stays all-ones for WMMA; store is guarded.
    int rowA = rowBase + m;
    if (rowA > n - 1) rowA = n - 1;
    const unsigned short* xrow = Xbf + (size_t)rowA * D;

    v8f acc[8] = {};

    union AFrag { uint4 q[2]; v16bf v; };

    #pragma unroll
    for (int ks = 0; ks < 4; ++ks) {
        // A fragment (16x32 bf16): lane holds K = {base..base+15},
        // base = half*16, within this 32-wide K step. Two contiguous 16B runs.
        const unsigned short* pa = xrow + ks * 32 + half * 16;
        AFrag af;
        af.q[0] = *(const uint4*)(pa);        // K = base .. base+7
        af.q[1] = *(const uint4*)(pa + 8);    // K = base+8 .. base+15

        const unsigned short* pb = Wz + ((size_t)ks * 8) * 512 + lane * 16;
        #pragma unroll
        for (int nt = 0; nt < 8; ++nt) {
            v16bf b = *(const v16bf*)(pb + nt * 512);   // 32B, already in WMMA layout
            acc[nt] = __builtin_amdgcn_wmma_f32_16x16x32_bf16(
                false, af.v, false, b, (short)0, acc[nt], false, false);
        }
    }

    // store D: VGPR r, lanes 0-15 -> row r; lanes 16-31 -> row 8+r
    #pragma unroll
    for (int nt = 0; nt < 8; ++nt) {
        #pragma unroll
        for (int r = 0; r < 8; ++r) {
            int rr = rowBase + half * 8 + r;
            if (rr < n) H[(size_t)rr * D + nt * 16 + m] = acc[nt][r];
        }
    }
}

// ---------------------------------------------------------------- AGG init (self-loop + bias)
__global__ __launch_bounds__(256) void init_agg(const float* __restrict__ H,
                                                const float* __restrict__ dinv,
                                                const float* __restrict__ bias,
                                                float* __restrict__ AGG, int n) {
    int i = blockIdx.x * blockDim.x + threadIdx.x;   // one float4 chunk each
    int node = i >> 5;
    int c = (i & 31) * 4;
    if (node >= n) return;
    float di = dinv[node];
    float nn = di * di;
    float4 h = *(const float4*)(H + (size_t)node * D + c);
    float4 b = *(const float4*)(bias + c);
    float4 o;
    o.x = h.x * nn + b.x; o.y = h.y * nn + b.y;
    o.z = h.z * nn + b.z; o.w = h.w * nn + b.w;
    *(float4*)(AGG + (size_t)node * D + c) = o;
}

// ---------------------------------------------------------------- edge scatter-add
__global__ __launch_bounds__(256) void scatter_kernel(const float* __restrict__ H,
                                                      const int* __restrict__ src,
                                                      const int* __restrict__ dst,
                                                      const float* __restrict__ dinv,
                                                      float* __restrict__ AGG, int E) {
    int wid  = (blockIdx.x * blockDim.x + threadIdx.x) >> 5;
    int lane = threadIdx.x & 31;
    if (wid >= E) return;
    if (wid + 4096 < E) {
        __builtin_prefetch(&src[wid + 4096], 0, 1);
        __builtin_prefetch(&dst[wid + 4096], 0, 1);
    }
    int s = src[wid];
    int d = dst[wid];
    float nrm = dinv[s] * dinv[d];
    float4 h = *(const float4*)(H + (size_t)s * D + lane * 4);
    float* out = AGG + (size_t)d * D + lane * 4;
    atomicAdd(out + 0, h.x * nrm);
    atomicAdd(out + 1, h.y * nrm);
    atomicAdd(out + 2, h.z * nrm);
    atomicAdd(out + 3, h.w * nrm);
}

// ---------------------------------------------------------------- ELU + JK max + next Xbf
__global__ __launch_bounds__(256) void finish_kernel(const float* __restrict__ AGG,
                                                     unsigned short* __restrict__ Xbf,
                                                     float* __restrict__ JK,
                                                     int n, int apply_elu) {
    int i = blockIdx.x * blockDim.x + threadIdx.x;
    int node = i >> 5;
    int c = (i & 31) * 4;
    if (node >= n) return;
    size_t off = (size_t)node * D + c;
    float4 a = *(const float4*)(AGG + off);
    float4 h = a;
    if (apply_elu) {
        h.x = (a.x > 0.0f) ? a.x : (__expf(a.x) - 1.0f);
        h.y = (a.y > 0.0f) ? a.y : (__expf(a.y) - 1.0f);
        h.z = (a.z > 0.0f) ? a.z : (__expf(a.z) - 1.0f);
        h.w = (a.w > 0.0f) ? a.w : (__expf(a.w) - 1.0f);
    }
    float4 j = *(const float4*)(JK + off);
    j.x = fmaxf(j.x, h.x); j.y = fmaxf(j.y, h.y);
    j.z = fmaxf(j.z, h.z); j.w = fmaxf(j.w, h.w);
    *(float4*)(JK + off) = j;
    // next-layer activations in bf16, row-major (A-fragment friendly)
    uint2 o;
    o.x = (unsigned)f2bf(h.x) | ((unsigned)f2bf(h.y) << 16);
    o.y = (unsigned)f2bf(h.z) | ((unsigned)f2bf(h.w) << 16);
    *(uint2*)(Xbf + off) = o;
}

// ---------------------------------------------------------------- launcher
static inline size_t align256(size_t x) { return (x + 255) & ~(size_t)255; }

extern "C" void kernel_launch(void* const* d_in, const int* in_sizes, int n_in,
                              void* d_out, int out_size, void* d_ws, size_t ws_size,
                              hipStream_t stream) {
    const float* x   = (const float*)d_in[0];
    const int*   ei  = (const int*)d_in[1];
    const float* W0  = (const float*)d_in[2];
    const float* b0  = (const float*)d_in[3];
    const float* Ws  = (const float*)d_in[4];
    const float* bs  = (const float*)d_in[5];
    float* out = (float*)d_out;

    const int n = in_sizes[0] / D;          // 50000
    const int E = in_sizes[1] / 2;          // 800000
    const int* srcp = ei;
    const int* dstp = ei + E;

    // workspace layout (256B-aligned sections)
    char* base = (char*)d_ws;
    size_t off = 0;
    unsigned short* Xbf = (unsigned short*)(base + off); off = align256(off + (size_t)n * D * 2);
    float* H    = (float*)(base + off); off = align256(off + (size_t)n * D * 4);
    float* AGG  = (float*)(base + off); off = align256(off + (size_t)n * D * 4);
    float* dinv = (float*)(base + off); off = align256(off + (size_t)n * 4);
    unsigned short* Wz = (unsigned short*)(base + off);   // 16384 bf16 = 32KB

    const int T = 256;
    // degree (self-loop => start at 1.0) and symmetric norm
    fill_kernel<<<(n + T - 1) / T, T, 0, stream>>>(dinv, 1.0f, n);
    deg_kernel<<<(E + T - 1) / T, T, 0, stream>>>(dstp, dinv, E);
    dinv_kernel<<<(n + T - 1) / T, T, 0, stream>>>(dinv, n);
    // JK max accumulator
    fill_kernel<<<(out_size + T - 1) / T, T, 0, stream>>>(out, -INFINITY, out_size);
    // layer-0 activations -> bf16
    const int total4 = n * D / 4;
    x2bf_kernel<<<(total4 + T - 1) / T, T, 0, stream>>>(x, Xbf, total4);

    const int chunkThreads = n * 32;
    const int cgrid = (chunkThreads + T - 1) / T;
    const int ggrid = (n + 127) / 128;
    const int sgrid = (E * 32 + T - 1) / T;

    for (int l = 0; l < 8; ++l) {
        const float* Wl = (l == 0) ? W0 : Ws + (size_t)(l - 1) * D * D;
        const float* bl = (l == 0) ? b0 : bs + (size_t)(l - 1) * D;

        swizzleW_kernel<<<4, T, 0, stream>>>(Wl, Wz);
        gemm_wmma<<<ggrid, T, 0, stream>>>(Xbf, Wz, H, n);
        init_agg<<<cgrid, T, 0, stream>>>(H, dinv, bl, AGG, n);
        scatter_kernel<<<sgrid, T, 0, stream>>>(H, srcp, dstp, dinv, AGG, E);
        finish_kernel<<<cgrid, T, 0, stream>>>(AGG, Xbf, out, n, (l < 7) ? 1 : 0);
    }
}
